// GraphormerEncoder_81793357185845
// MI455X (gfx1250) — matrix operand order repaired and checked by reading, not compiled, not run on hardware
//
#include <hip/hip_runtime.h>
#include <hip/hip_bf16.h>
#include <cstdint>

// Problem dims (fixed by the reference)
#define Bb 4
#define Nn 384
#define Ee 768
#define Hh 32
#define Ff 3072
#define Ll 12
#define Dd 24
#define Tt (Nn * Bb)   // 1536 tokens

typedef __attribute__((ext_vector_type(16))) __bf16 v16bf;
typedef __attribute__((ext_vector_type(8)))  float  v8f;

enum { OUT_BF16 = 1, OUT_F32 = 2, DO_SCALE = 4, DO_GELU = 8 };

__device__ __forceinline__ v8f wmma_bf16(v16bf a, v16bf b, v8f c) {
  // D = A(16x32 bf16) x B(32x16 bf16) + C(16x16 f32)
  return __builtin_amdgcn_wmma_f32_16x16x32_bf16(false, a, false, b, (short)0, c,
                                                 false, false);
}

__device__ __forceinline__ v8f zero_v8f() {
  v8f z;
#pragma unroll
  for (int i = 0; i < 8; ++i) z[i] = 0.0f;
  return z;
}

// Async 16B global -> LDS copy (CDNA5 GLOBAL_LOAD_ASYNC_TO_LDS_B128, ASYNCcnt).
// Flat LDS addresses use addr[31:0] as the LDS offset (ISA 10.2), so the low
// 32 bits of a __shared__ pointer are the VDST LDS address.
__device__ __forceinline__ void async_copy16(void* lds, const void* gsrc) {
  asm volatile("global_load_async_to_lds_b128 %0, %1, off"
               :
               : "v"((unsigned)(size_t)lds),
                 "v"((unsigned long long)(size_t)gsrc)
               : "memory");
}
__device__ __forceinline__ void wait_async() {
  asm volatile("s_wait_asynccnt 0" ::: "memory");
}

// A fragment (16x32, 16-bit): lanes 0-15 row M=lane get K0-15 contiguous,
// lanes 16-31 same rows get K16-31 (matches load_b_frag k order).
__device__ __forceinline__ v16bf load_a_frag(const __bf16* as, int lda,
                                             int row_base, int koff, int lane) {
  const int r  = row_base + (lane & 15);
  const int kh = (lane >> 4) << 4;  // 0 or 16
  const __bf16* p = as + r * lda + koff;
  v16bf a;
#pragma unroll
  for (int i = 0; i < 8; ++i) a[i] = p[kh + i];
#pragma unroll
  for (int i = 0; i < 8; ++i) a[8 + i] = p[kh + 8 + i];
  return a;
}

// B fragment (32x16, 16-bit) from N-major LDS tile Bt[col][k]:
// lanes 0-15 col N=lane get K0-15 contiguous, lanes 16-31 get K16-31.
__device__ __forceinline__ v16bf load_b_frag(const __bf16* bt, int ldb,
                                             int col_base, int koff, int lane) {
  const int c  = col_base + (lane & 15);
  const int kh = (lane >> 4) << 4;  // 0 or 16
  const __bf16* p = bt + c * ldb + koff + kh;
  v16bf b;
#pragma unroll
  for (int i = 0; i < 16; ++i) b[i] = p[i];
  return b;
}

__device__ __forceinline__ float gelu_tanh(float x) {
  const float x3 = x * x * x;
  return 0.5f * x * (1.0f + tanhf(0.7978845608028654f * (x + 0.044715f * x3)));
}

// ---------------------------------------------------------------------------
// Generic GEMM: Y[M,NC] = epilogue( Xb[M,K](bf16) @ W[K,NC](f32->bf16) + bias )
// Block tile 128x128, K-step 32, double-buffered LDS; 8 waves x (32x64) each.
// A-tile fills use async global->LDS (no VGPR staging); B-tile needs the
// f32->bf16 convert+transpose so it stays through VGPRs.
// ---------------------------------------------------------------------------
template <int FLAGS>
__global__ __launch_bounds__(256) void gemm_bf16_kernel(
    const __bf16* __restrict__ X, const float* __restrict__ W,
    const float* __restrict__ bias, float scale, float* __restrict__ Yf,
    __bf16* __restrict__ Yb, int M, int K, int NC) {
  __shared__ __bf16 As[2][128][40];  // [stage][row][k], padded (bank-safe)
  __shared__ __bf16 Bt[2][128][40];  // [stage][col][k]
  const int tid  = threadIdx.x;
  const int lane = tid & 31;
  const int w    = tid >> 5;
  const int wr   = w & 3;            // 4 row groups of 32 rows
  const int wc   = w >> 2;           // 2 col groups of 64 cols
  const int n0   = blockIdx.x * 128;
  const int m0   = blockIdx.y * 128;

  auto load_tiles = [&](int kt, int pp) {
    // A tile: 128x32 bf16, async 16B copies (2 per thread)
#pragma unroll
    for (int i = 0; i < 2; ++i) {
      const int idx = tid + i * 256;
      const int row = idx >> 2;
      const int ch  = (idx & 3) << 3;
      async_copy16(&As[pp][row][ch], X + (size_t)(m0 + row) * K + kt + ch);
    }
    // B tile: 32x128 f32 -> bf16, transposed into Bt[col][k]
#pragma unroll
    for (int i = 0; i < 4; ++i) {
      const int idx = tid + i * 256;
      const int kr  = idx >> 5;
      const int ch  = (idx & 31) << 2;
      const float* wsrc = W + (size_t)(kt + kr) * NC + n0 + ch;
      const float4 f = *(const float4*)wsrc;
      Bt[pp][ch + 0][kr] = (__bf16)f.x;
      Bt[pp][ch + 1][kr] = (__bf16)f.y;
      Bt[pp][ch + 2][kr] = (__bf16)f.z;
      Bt[pp][ch + 3][kr] = (__bf16)f.w;
      if (kt + 32 < K) __builtin_prefetch(wsrc + (size_t)32 * NC, 0, 0);
    }
  };

  v8f acc[2][4];
#pragma unroll
  for (int r = 0; r < 2; ++r)
#pragma unroll
    for (int c = 0; c < 4; ++c) acc[r][c] = zero_v8f();

  load_tiles(0, 0);
  wait_async();
  __syncthreads();
  int p = 0;
  for (int kt = 0; kt < K; kt += 32) {
    if (kt + 32 < K) load_tiles(kt + 32, p ^ 1);  // overlap fill of next stage
    const v16bf a0 = load_a_frag(&As[p][0][0], 40, wr * 32, 0, lane);
    const v16bf a1 = load_a_frag(&As[p][0][0], 40, wr * 32 + 16, 0, lane);
#pragma unroll
    for (int c = 0; c < 4; ++c) {
      const v16bf bf = load_b_frag(&Bt[p][0][0], 40, wc * 64 + c * 16, 0, lane);
      acc[0][c] = wmma_bf16(a0, bf, acc[0][c]);
      acc[1][c] = wmma_bf16(a1, bf, acc[1][c]);
    }
    wait_async();      // async fills of stage p^1 done before the barrier
    __syncthreads();
    p ^= 1;
  }

  // C layout: VGPR v -> row v (lanes 0-15, col=lane) / row v+8 (lanes 16-31)
  const int rofs = (lane >> 4) << 3;
  const int cofs = lane & 15;
#pragma unroll
  for (int r = 0; r < 2; ++r) {
#pragma unroll
    for (int c = 0; c < 4; ++c) {
      const int colg = n0 + wc * 64 + c * 16 + cofs;
      const float bv = bias[colg];
#pragma unroll
      for (int v = 0; v < 8; ++v) {
        const int rowg = m0 + wr * 32 + r * 16 + v + rofs;
        float val = acc[r][c][v] + bv;
        if (FLAGS & DO_SCALE) val *= scale;
        if (FLAGS & DO_GELU) val = gelu_tanh(val);
        const size_t oi = (size_t)rowg * NC + colg;
        if (FLAGS & OUT_F32) Yf[oi] = val;
        if (FLAGS & OUT_BF16) Yb[oi] = (__bf16)val;
      }
    }
  }
}

// ---------------------------------------------------------------------------
// Fused flash attention per (b,h): for a 128-row Q tile, loop over 128-col
// K/V tiles: S = Q K^T (WMMA) + bias, online softmax (fp32), oacc += P @ V
// (WMMA, P bf16 via LDS).  q is pre-scaled by 1/sqrt(D) in the Q GEMM.
// Q/K d<24 chunks are async global->LDS copies; pad columns are zero stores.
// Padding mask is provably never active (node_type >= 1).
// ---------------------------------------------------------------------------
__global__ __launch_bounds__(256) void flash_attn_kernel(
    const __bf16* __restrict__ qb, const __bf16* __restrict__ kb,
    const __bf16* __restrict__ vb, const float* __restrict__ biasb,
    __bf16* __restrict__ ab) {
  __shared__ __bf16 Qs[128][40];    // [n-local][d]  (d padded 24->32)
  __shared__ __bf16 Ks[128][40];    // [m-local][d]
  __shared__ __bf16 Ps[128][136];   // [n-local][m-local] probs tile (bf16)
  __shared__ __bf16 Vs[32][136];    // [d][m-local]  (B-tile layout)
  const int tid = threadIdx.x, lane = tid & 31, w = tid >> 5;
  const int bh = blockIdx.x, b = bh >> 5, h = bh & 31;  // H == 32
  const int n0 = blockIdx.y * 128;
  const int rofs = (lane >> 4) << 3;
  const int cofs = lane & 15;

  // Q tile 128x32 (async copy d<24 chunks, zero the pad chunk)
#pragma unroll
  for (int i = 0; i < 2; ++i) {
    const int idx = tid + i * 256;
    const int row = idx >> 2, ch = (idx & 3) << 3;
    if (ch < Dd) {
      const size_t t = (size_t)(n0 + row) * Bb + b;
      async_copy16(&Qs[row][ch], qb + t * Ee + h * Dd + ch);
    } else {
      *(uint4*)&Qs[row][ch] = make_uint4(0u, 0u, 0u, 0u);
    }
  }

  float mrun[8], lsum[8];
  v8f oacc[2];
#pragma unroll
  for (int v = 0; v < 8; ++v) { mrun[v] = -3.0e38f; lsum[v] = 0.f; }
  oacc[0] = zero_v8f();
  oacc[1] = zero_v8f();

  for (int mt = 0; mt < Nn; mt += 128) {
    // K tile 128x32 (async copy d<24 chunks, zero pad)
#pragma unroll
    for (int i = 0; i < 2; ++i) {
      const int idx = tid + i * 256;
      const int row = idx >> 2, ch = (idx & 3) << 3;
      if (ch < Dd) {
        const size_t t = (size_t)(mt + row) * Bb + b;
        async_copy16(&Ks[row][ch], kb + t * Ee + h * Dd + ch);
      } else {
        *(uint4*)&Ks[row][ch] = make_uint4(0u, 0u, 0u, 0u);
      }
    }
    // V tile transposed: Vs[d][m_local], zero-pad d >= 24 (manual: transpose)
#pragma unroll
    for (int i = 0; i < 16; ++i) {
      const int idx = tid + i * 256;
      const int d = idx & 31, k = idx >> 5;
      __bf16 val = (__bf16)0.0f;
      if (d < Dd) val = vb[((size_t)(mt + k) * Bb + b) * Ee + h * Dd + d];
      Vs[d][k] = val;
    }
    wait_async();
    __syncthreads();

    // S tile: wave w owns rows n0+w*16 .. +15, all 128 cols (8 c-tiles)
    const v16bf aq = load_a_frag(&Qs[0][0], 40, w * 16, 0, lane);
    v8f s[8];
#pragma unroll
    for (int c = 0; c < 8; ++c) {
      const v16bf bf = load_b_frag(&Ks[0][0], 40, c * 16, 0, lane);
      s[c] = wmma_bf16(aq, bf, zero_v8f());
    }
    // + attention bias (batched loads per c-tile)
    const size_t brow0 =
        ((size_t)bh * Nn + n0 + w * 16 + rofs) * Nn + mt + cofs;
#pragma unroll
    for (int c = 0; c < 8; ++c) {
      float tb[8];
#pragma unroll
      for (int v = 0; v < 8; ++v) tb[v] = biasb[brow0 + (size_t)v * Nn + c * 16];
#pragma unroll
      for (int v = 0; v < 8; ++v) s[c][v] += tb[v];
    }
    // online softmax across this tile's 128 columns
    // (row v+rofs is striped over the 16 lanes of each half: xor 1..8 reduces)
#pragma unroll
    for (int v = 0; v < 8; ++v) {
      float mx = s[0][v];
#pragma unroll
      for (int c = 1; c < 8; ++c) mx = fmaxf(mx, s[c][v]);
#pragma unroll
      for (int off = 1; off <= 8; off <<= 1)
        mx = fmaxf(mx, __shfl_xor(mx, off, 32));
      const float mnew = fmaxf(mrun[v], mx);
      const float sc = expf(mrun[v] - mnew);
      float ps = 0.f;
#pragma unroll
      for (int c = 0; c < 8; ++c) {
        const float pr = expf(s[c][v] - mnew);
        s[c][v] = pr;
        ps += pr;
      }
#pragma unroll
      for (int off = 1; off <= 8; off <<= 1) ps += __shfl_xor(ps, off, 32);
      lsum[v] = lsum[v] * sc + ps;
      mrun[v] = mnew;
      oacc[0][v] *= sc;
      oacc[1][v] *= sc;
    }
    // P -> LDS as bf16 (each wave writes only its own 16 rows)
#pragma unroll
    for (int c = 0; c < 8; ++c)
#pragma unroll
      for (int v = 0; v < 8; ++v)
        Ps[w * 16 + v + rofs][c * 16 + cofs] = (__bf16)s[c][v];
    __syncthreads();

    // oacc += P @ V : K = 128 in 4 WMMA steps, 2 d-tiles
#pragma unroll
    for (int kk = 0; kk < 4; ++kk) {
      const v16bf ap = load_a_frag(&Ps[0][0], 136, w * 16, kk * 32, lane);
      const v16bf b0 = load_b_frag(&Vs[0][0], 136, 0, kk * 32, lane);
      const v16bf b1 = load_b_frag(&Vs[0][0], 136, 16, kk * 32, lane);
      oacc[0] = wmma_bf16(ap, b0, oacc[0]);
      oacc[1] = wmma_bf16(ap, b1, oacc[1]);
    }
    __syncthreads();
  }

  // normalize and store a[n,b,h,d] (bf16 for the Wo GEMM)
#pragma unroll
  for (int v = 0; v < 8; ++v) {
    const float inv = 1.0f / lsum[v];
    const int n = n0 + w * 16 + v + rofs;
    const size_t t = (size_t)n * Bb + b;
    ab[t * Ee + h * Dd + cofs] = (__bf16)(oacc[0][v] * inv);
    const int d1 = 16 + cofs;
    if (d1 < Dd) ab[t * Ee + h * Dd + d1] = (__bf16)(oacc[1][v] * inv);
  }
}

// Fused residual + LayerNorm; writes fp32 master and bf16 mirror.
__global__ __launch_bounds__(256) void residual_ln_kernel(
    const float* __restrict__ x, const float* __restrict__ delta,
    const float* __restrict__ gam, const float* __restrict__ bet,
    float* __restrict__ xout, __bf16* __restrict__ xbout) {
  __shared__ float ssum[8], ssq[8];
  const int tid = threadIdx.x, lane = tid & 31, w = tid >> 5;
  const size_t base = (size_t)blockIdx.x * Ee;
  float v[3];
  float s = 0.f, q = 0.f;
#pragma unroll
  for (int i = 0; i < 3; ++i) {
    const int e = tid + i * 256;
    v[i] = x[base + e] + delta[base + e];
    s += v[i];
    q += v[i] * v[i];
  }
#pragma unroll
  for (int off = 16; off > 0; off >>= 1) {
    s += __shfl_xor(s, off, 32);
    q += __shfl_xor(q, off, 32);
  }
  if (lane == 0) { ssum[w] = s; ssq[w] = q; }
  __syncthreads();
  float S = 0.f, Q = 0.f;
#pragma unroll
  for (int i = 0; i < 8; ++i) { S += ssum[i]; Q += ssq[i]; }
  const float mean = S * (1.0f / Ee);
  const float var  = Q * (1.0f / Ee) - mean * mean;
  const float rstd = rsqrtf(var + 1e-5f);
#pragma unroll
  for (int i = 0; i < 3; ++i) {
    const int e = tid + i * 256;
    const float y = (v[i] - mean) * rstd * gam[e] + bet[e];
    xout[base + e] = y;
    xbout[base + e] = (__bf16)y;
  }
}

// Fused embedding gathers. node_type in [1,S] so the padding mask is never hit.
__global__ __launch_bounds__(256) void embed_kernel(
    const int* __restrict__ node_type, const float* __restrict__ feat,
    const int* __restrict__ in_deg, const int* __restrict__ out_deg,
    const float* __restrict__ node_emb, const float* __restrict__ ind_emb,
    const float* __restrict__ outd_emb, float* __restrict__ x,
    __bf16* __restrict__ xb) {
  const int bn = blockIdx.x;
  const int b = bn / Nn, n = bn % Nn;
  const int nt = node_type[bn];
  const int id = in_deg[bn];
  const int od = out_deg[bn];
  const size_t dst  = ((size_t)n * Bb + b) * Ee;  // x is [N,B,E]
  const size_t srcf = (size_t)bn * Ee;            // feat is [B,N,E]
#pragma unroll
  for (int i = 0; i < 3; ++i) {
    const int e = threadIdx.x + i * 256;
    const float val = node_emb[(size_t)nt * Ee + e] + ind_emb[(size_t)id * Ee + e] +
                      outd_emb[(size_t)od * Ee + e] + feat[srcf + e];
    x[dst + e] = val;
    xb[dst + e] = (__bf16)val;
  }
}

// Precompute attention bias [B,H,N,N] (reused by all 12 layers).
__global__ __launch_bounds__(128) void bias_kernel(
    const float* __restrict__ attn_bias, const int* __restrict__ spatial_pos,
    const float* __restrict__ sp_emb, const float* __restrict__ sp_emb_rev,
    float* __restrict__ biasb) {
  const int bn = blockIdx.x;
  const int b = bn / Nn, n = bn % Nn;
  const int m = blockIdx.y * 128 + threadIdx.x;
  const int sp  = spatial_pos[(size_t)bn * Nn + m];
  const int spr = spatial_pos[((size_t)b * Nn + m) * Nn + n];
  const float gb = attn_bias[(size_t)bn * Nn + m];
#pragma unroll
  for (int h = 0; h < Hh; ++h) {
    biasb[(((size_t)b * Hh + h) * Nn + n) * Nn + m] =
        sp_emb[sp * Hh + h] + sp_emb_rev[spr * Hh + h] + gb;
  }
}

// ---------------------------------------------------------------------------
extern "C" void kernel_launch(void* const* d_in, const int* in_sizes, int n_in,
                              void* d_out, int out_size, void* d_ws,
                              size_t ws_size, hipStream_t stream) {
  (void)in_sizes; (void)n_in; (void)out_size; (void)ws_size;
  const int*   node_type = (const int*)d_in[0];
  const float* feat      = (const float*)d_in[1];
  const int*   in_deg    = (const int*)d_in[2];
  const int*   out_deg   = (const int*)d_in[3];
  const float* attn_bias = (const float*)d_in[4];
  const int*   spatial   = (const int*)d_in[5];
  const float* node_emb  = (const float*)d_in[6];
  const float* ind_emb   = (const float*)d_in[7];
  const float* outd_emb  = (const float*)d_in[8];
  const float* sp_emb    = (const float*)d_in[9];
  const float* sp_embr   = (const float*)d_in[10];
  const float* Wq = (const float*)d_in[11]; const float* bq = (const float*)d_in[12];
  const float* Wk = (const float*)d_in[13]; const float* bk = (const float*)d_in[14];
  const float* Wv = (const float*)d_in[15]; const float* bv = (const float*)d_in[16];
  const float* Wo = (const float*)d_in[17]; const float* bo = (const float*)d_in[18];
  const float* W1 = (const float*)d_in[19]; const float* b1 = (const float*)d_in[20];
  const float* W2 = (const float*)d_in[21]; const float* b2 = (const float*)d_in[22];
  const float* ln1s = (const float*)d_in[23]; const float* ln1b = (const float*)d_in[24];
  const float* ln2s = (const float*)d_in[25]; const float* ln2b = (const float*)d_in[26];

  char* ws = (char*)d_ws;
  size_t off = 0;
  auto take = [&](size_t bytes) -> void* {
    void* p = ws + off;
    off += (bytes + 255) & ~(size_t)255;
    return p;
  };
  float*  x     = (float*)take((size_t)Tt * Ee * 4);
  float*  delta = (float*)take((size_t)Tt * Ee * 4);
  __bf16* xb    = (__bf16*)take((size_t)Tt * Ee * 2);
  __bf16* qb    = (__bf16*)take((size_t)Tt * Ee * 2);
  __bf16* kb    = (__bf16*)take((size_t)Tt * Ee * 2);
  __bf16* vbuf  = (__bf16*)take((size_t)Tt * Ee * 2);
  __bf16* abuf  = (__bf16*)take((size_t)Tt * Ee * 2);
  __bf16* hb    = (__bf16*)take((size_t)Tt * Ff * 2);
  float*  biasb = (float*)take((size_t)Bb * Hh * Nn * Nn * 4);

  embed_kernel<<<Bb * Nn, 256, 0, stream>>>(node_type, feat, in_deg, out_deg,
                                            node_emb, ind_emb, outd_emb, x, xb);
  bias_kernel<<<dim3(Bb * Nn, Nn / 128), 128, 0, stream>>>(attn_bias, spatial,
                                                           sp_emb, sp_embr, biasb);

  const float scaling = 0.20412414523193154f;  // 1/sqrt(D), D=24
  const dim3 gE(Ee / 128, Tt / 128);
  const dim3 gF(Ff / 128, Tt / 128);
  const dim3 gA(Bb * Hh, Nn / 128);

  for (int l = 0; l < Ll; ++l) {
    const float* Wq_l = Wq + (size_t)l * Ee * Ee;
    const float* Wk_l = Wk + (size_t)l * Ee * Ee;
    const float* Wv_l = Wv + (size_t)l * Ee * Ee;
    const float* Wo_l = Wo + (size_t)l * Ee * Ee;
    const float* W1_l = W1 + (size_t)l * Ee * Ff;
    const float* W2_l = W2 + (size_t)l * Ff * Ee;
    const float* bq_l = bq + (size_t)l * Ee;
    const float* bk_l = bk + (size_t)l * Ee;
    const float* bv_l = bv + (size_t)l * Ee;
    const float* bo_l = bo + (size_t)l * Ee;
    const float* b1_l = b1 + (size_t)l * Ff;
    const float* b2_l = b2 + (size_t)l * Ee;
    const float* s1 = ln1s + (size_t)l * Ee; const float* t1 = ln1b + (size_t)l * Ee;
    const float* s2 = ln2s + (size_t)l * Ee; const float* t2 = ln2b + (size_t)l * Ee;

    gemm_bf16_kernel<OUT_BF16 | DO_SCALE><<<gE, 256, 0, stream>>>(
        xb, Wq_l, bq_l, scaling, nullptr, qb, Tt, Ee, Ee);
    gemm_bf16_kernel<OUT_BF16><<<gE, 256, 0, stream>>>(
        xb, Wk_l, bk_l, 1.f, nullptr, kb, Tt, Ee, Ee);
    gemm_bf16_kernel<OUT_BF16><<<gE, 256, 0, stream>>>(
        xb, Wv_l, bv_l, 1.f, nullptr, vbuf, Tt, Ee, Ee);

    flash_attn_kernel<<<gA, 256, 0, stream>>>(qb, kb, vbuf, biasb, abuf);

    gemm_bf16_kernel<OUT_F32><<<gE, 256, 0, stream>>>(
        abuf, Wo_l, bo_l, 1.f, delta, nullptr, Tt, Ee, Ee);
    residual_ln_kernel<<<Tt, 256, 0, stream>>>(x, delta, s1, t1, x, xb);

    gemm_bf16_kernel<OUT_BF16 | DO_GELU><<<gF, 256, 0, stream>>>(
        xb, W1_l, b1_l, 1.f, nullptr, hb, Tt, Ee, Ff);
    gemm_bf16_kernel<OUT_F32><<<gE, 256, 0, stream>>>(
        hb, W2_l, b2_l, 1.f, delta, nullptr, Tt, Ff, Ee);

    float* xo = (l == Ll - 1) ? (float*)d_out : x;  // final LN -> d_out [N,B,E]
    residual_ln_kernel<<<Tt, 256, 0, stream>>>(x, delta, s2, t2, xo, xb);
  }
}
